// CoarseSelfAttention_2963527434299
// MI455X (gfx1250) — matrix operand order repaired and checked by reading, not compiled, not run on hardware
//
#include <hip/hip_runtime.h>
#include <stdint.h>

// ---------------------------------------------------------------------------
// CoarseSelfAttention for MI455X (gfx1250, wave32, WMMA bf16 16x16x32)
// ---------------------------------------------------------------------------

#define BATCH 16
#define NH    4
#define HD    64
#define NSEQ  1024   // 32*32 after downsample
#define KTOP  256

typedef __attribute__((ext_vector_type(16))) __bf16 v16bf;
typedef __attribute__((ext_vector_type(8)))  float  v8f;

union V16 { v16bf v; uint4 q[2]; unsigned short u[16]; };
union PK8 { uint4 q; unsigned short u[8]; };

static __device__ __forceinline__ unsigned short f2bf(float f) {
  unsigned int u = __float_as_uint(f);
  u += 0x7FFFu + ((u >> 16) & 1u);            // round-to-nearest-even
  return (unsigned short)(u >> 16);
}
static __device__ __forceinline__ v8f vzero() {
  v8f z;
#pragma unroll
  for (int i = 0; i < 8; ++i) z[i] = 0.f;
  return z;
}
static __device__ __forceinline__ v8f wmma_bf16(const V16& a, const V16& b, v8f c) {
  return __builtin_amdgcn_wmma_f32_16x16x32_bf16(false, a.v, false, b.v, (short)0, c,
                                                 false, false);
}

// ---------------------------------------------------------------------------
// Prep kernels (tiny, one-shot per launch)
// ---------------------------------------------------------------------------

__global__ __launch_bounds__(256) void k_cvt(const float* __restrict__ src,
                                             unsigned short* __restrict__ dst, int n) {
  int i = blockIdx.x * 256 + threadIdx.x;
  if (i < n) dst[i] = f2bf(src[i]);
}

// qkv_w [192][64] -> wT [64][192] bf16
__global__ __launch_bounds__(256) void k_prep_qkvw(const float* __restrict__ w,
                                                   unsigned short* __restrict__ dst) {
  int i = blockIdx.x * 256 + threadIdx.x;
  if (i < 64 * 192) {
    int d = i / 192, e = i % 192;
    dst[i] = f2bf(w[e * 64 + d]);
  }
}

// conv_up_w [ic][oc][4][4] -> 4 parity-class GEMM A matrices [cls][oc][ic*4+s*2+t]
__global__ __launch_bounds__(256) void k_prep_wu(const float* __restrict__ w,
                                                 unsigned short* __restrict__ dst) {
  int i = blockIdx.x * 256 + threadIdx.x;
  if (i >= 4 * 256 * 1024) return;
  int cls = i >> 18;
  int r   = i & 262143;
  int oc  = r >> 10;
  int k   = r & 1023;
  int ic  = k >> 2, s = (k >> 1) & 1, t = k & 1;
  int po  = cls >> 1, pw = cls & 1;
  int kh  = po ? (s ? 2 : 0) : (s ? 3 : 1);
  int kw  = pw ? (t ? 2 : 0) : (t ? 3 : 1);
  dst[i] = f2bf(w[(ic * 256 + oc) * 16 + kh * 4 + kw]);
}

// ---------------------------------------------------------------------------
// Conv down: implicit GEMM  M=256(oc) x K=4096(ic,kh,kw) x N=1024(spatial)/batch
// block = 256 thr (8 waves) -> 128(M) x 128(N) tile, K chunked by 32,
// double-buffered LDS with software-pipelined gathers (branchless clamp+select)
// ---------------------------------------------------------------------------

__global__ __launch_bounds__(256) void k_conv_down(
    const float* __restrict__ x,            // [16][256][64][64] f32
    const unsigned short* __restrict__ wd,  // [256][4096] bf16
    const float* __restrict__ bias,         // [256]
    unsigned short* __restrict__ y)         // [16][4][1024][64] bf16
{
  __shared__ __align__(16) unsigned short As[2][128 * 40];   // padded pitch 40
  __shared__ __align__(16) unsigned short Bs[2][32 * 136];   // padded pitch 136
  const int b = blockIdx.z;
  const int nbase = blockIdx.y * 128;
  const int mbase = blockIdx.x * 128;
  const int tid = threadIdx.x;
  const int lane = tid & 31, wave = tid >> 5;
  const int hi = lane >> 4, lo = lane & 15;

  v8f acc[8];
#pragma unroll
  for (int i = 0; i < 8; ++i) acc[i] = vzero();

  const int arow = tid >> 1, acol = (tid & 1) * 16;
  const int brow = tid >> 3, bcol0 = (tid & 7) * 16;

  uint4 a0, a1;
  float fv[16];
  unsigned okm;

  auto gather = [&](int kc) {
    const uint4* asrc = (const uint4*)(wd + (size_t)(mbase + arow) * 4096 + kc * 32 + acol);
    a0 = asrc[0];
    a1 = asrc[1];
    int kg = kc * 32 + brow;
    int ic = kg >> 4, kh = (kg >> 2) & 3, kw = kg & 3;
    const float* xp = x + (size_t)(b * 256 + ic) * 4096;
    okm = 0u;
#pragma unroll
    for (int j = 0; j < 16; ++j) {
      int n = nbase + bcol0 + j;
      int oh = n >> 5, ow = n & 31;
      int h = 2 * oh - 1 + kh, w = 2 * ow - 1 + kw;
      unsigned ok = ((unsigned)h < 64u) & ((unsigned)w < 64u);
      int hc = ok ? h : 0, wc = ok ? w : 0;
      fv[j] = xp[hc * 64 + wc];       // always-legal load, masked at stage
      okm |= ok << j;
    }
  };
  auto stage = [&](int buf) {
    uint4* ad = (uint4*)(As[buf] + arow * 40 + acol);
    ad[0] = a0;
    ad[1] = a1;
    PK8 p0, p1;
#pragma unroll
    for (int j = 0; j < 8; ++j) {
      p0.u[j] = ((okm >> j) & 1u) ? f2bf(fv[j]) : (unsigned short)0;
      p1.u[j] = ((okm >> (j + 8)) & 1u) ? f2bf(fv[j + 8]) : (unsigned short)0;
    }
    *(uint4*)(Bs[buf] + brow * 136 + bcol0) = p0.q;
    *(uint4*)(Bs[buf] + brow * 136 + bcol0 + 8) = p1.q;
  };

  gather(0);
  stage(0);
  for (int kc = 0; kc < 128; ++kc) {
    __syncthreads();
    const int cur = kc & 1;
    if (kc + 1 < 128) gather(kc + 1);     // loads in flight during WMMAs
    V16 af;
    {
      int m = wave * 16 + lo;
      af.q[0] = *(const uint4*)(As[cur] + m * 40 + hi * 8);
      af.q[1] = *(const uint4*)(As[cur] + m * 40 + 16 + hi * 8);
    }
#pragma unroll
    for (int nt = 0; nt < 8; ++nt) {
      V16 bf;
      for (int s = 0; s < 16; ++s) bf.u[s] = Bs[cur][((s >> 3) * 16 + hi * 8 + (s & 7)) * 136 + nt * 16 + lo];
      acc[nt] = wmma_bf16(af, bf, acc[nt]);
    }
    if (kc + 1 < 128) stage(1 - cur);
  }
#pragma unroll
  for (int nt = 0; nt < 8; ++nt) {
    int n = nbase + nt * 16 + lo;
#pragma unroll
    for (int v = 0; v < 8; ++v) {
      int oc = mbase + wave * 16 + v + 8 * hi;
      float val = acc[nt][v] + bias[oc];
      int head = oc >> 6, d = oc & 63;
      y[((size_t)(b * 4 + head) * 1024 + n) * 64 + d] = f2bf(val);
    }
  }
}

// ---------------------------------------------------------------------------
// QKV: per (b,h): [1024 x 64] @ [64 x 192] + bias, split into q/k/v bf16
// ---------------------------------------------------------------------------

__global__ __launch_bounds__(256) void k_qkv(
    const unsigned short* __restrict__ y,   // [bh][1024][64] bf16
    const unsigned short* __restrict__ wt,  // [64][192] bf16 (transposed qkv_w)
    const float* __restrict__ qkvb,         // [192]
    unsigned short* __restrict__ qb, unsigned short* __restrict__ kb,
    unsigned short* __restrict__ vb)
{
  const int bh = blockIdx.y;
  const int tid = threadIdx.x, lane = tid & 31, wave = tid >> 5;
  const int hi = lane >> 4, lo = lane & 15;
  const int rowbase = blockIdx.x * 128 + wave * 16;
  const unsigned short* yp = y + (size_t)bh * 65536;

  V16 af[2];
#pragma unroll
  for (int kc = 0; kc < 2; ++kc) {
    const unsigned short* p = yp + (size_t)(rowbase + lo) * 64 + kc * 32 + hi * 8;
    af[kc].q[0] = *(const uint4*)p;
    af[kc].q[1] = *(const uint4*)(p + 16);
  }
  for (int et = 0; et < 12; ++et) {
    float bv = qkvb[et * 16 + lo];
    v8f c;
#pragma unroll
    for (int v = 0; v < 8; ++v) c[v] = bv;
#pragma unroll
    for (int kc = 0; kc < 2; ++kc) {
      V16 bf;
      for (int s = 0; s < 16; ++s) bf.u[s] = wt[(size_t)(kc * 32 + (s >> 3) * 16 + hi * 8 + (s & 7)) * 192 + et * 16 + lo];
      c = wmma_bf16(af[kc], bf, c);
    }
    int e = et * 16 + lo;
    unsigned short* dst = (e < 64) ? qb : ((e < 128) ? kb : vb);
    dst += (size_t)bh * 65536 + (e & 63);
#pragma unroll
    for (int v = 0; v < 8; ++v) {
      int n = rowbase + v + 8 * hi;
      dst[(size_t)n * 64] = f2bf(c[v]);
    }
  }
}

// ---------------------------------------------------------------------------
// Attention: attn = softmax(K @ Q^T) rows, O = attn @ V, column-sum partials.
// Two-pass flash style per (bh, 128-row block). Deterministic colsum partials.
// ---------------------------------------------------------------------------

__global__ __launch_bounds__(256) void k_attn(
    const unsigned short* __restrict__ qb, const unsigned short* __restrict__ kb,
    const unsigned short* __restrict__ vb,
    float* __restrict__ cspart,             // [bh*8+iblk][8 waves][1024]
    unsigned short* __restrict__ z)         // [16][256][1024] bf16
{
  __shared__ __align__(16) unsigned short Qt[64 * 72];       // Q^T staged [d][j]
  __shared__ __align__(16) unsigned short Ps[8][16 * 40];    // per-wave P tile
  const int bh = blockIdx.y;
  const int tid = threadIdx.x, lane = tid & 31, wave = tid >> 5;
  const int hi = lane >> 4, lo = lane & 15;
  const int i0 = blockIdx.x * 128 + wave * 16;
  const unsigned short* qp = qb + (size_t)bh * 65536;
  const unsigned short* kp = kb + (size_t)bh * 65536;
  const unsigned short* vp = vb + (size_t)bh * 65536;

  V16 af[2];
#pragma unroll
  for (int kc = 0; kc < 2; ++kc) {
    const unsigned short* p = kp + (size_t)(i0 + lo) * 64 + kc * 32 + hi * 8;
    af[kc].q[0] = *(const uint4*)p;
    af[kc].q[1] = *(const uint4*)(p + 16);
  }

  float vm[8], vl[8];
#pragma unroll
  for (int v = 0; v < 8; ++v) { vm[v] = -__builtin_inff(); vl[v] = 0.f; }

  const int qj = tid >> 2, qd0 = (tid & 3) * 16;

  // ---- pass 1: row max / sum-exp ----
  for (int jc = 0; jc < 16; ++jc) {
    {
      V16 tmp;
      const unsigned short* p = qp + (size_t)(jc * 64 + qj) * 64 + qd0;
      tmp.q[0] = *(const uint4*)p;
      tmp.q[1] = *(const uint4*)(p + 8);
#pragma unroll
      for (int u = 0; u < 16; ++u) Qt[(qd0 + u) * 72 + qj] = tmp.u[u];
    }
    __syncthreads();
#pragma unroll
    for (int js = 0; js < 4; ++js) {
      v8f s = vzero();
#pragma unroll
      for (int kc = 0; kc < 2; ++kc) {
        V16 bf;
        for (int s = 0; s < 16; ++s) bf.u[s] = Qt[(kc * 32 + (s >> 3) * 16 + hi * 8 + (s & 7)) * 72 + js * 16 + lo];
        s = wmma_bf16(af[kc], bf, s);
      }
#pragma unroll
      for (int v = 0; v < 8; ++v) {
        float xv = s[v];
        if (xv > vm[v]) { vl[v] = vl[v] * __expf(vm[v] - xv) + 1.f; vm[v] = xv; }
        else             vl[v] += __expf(xv - vm[v]);
      }
    }
    __syncthreads();
  }
  // butterfly combine across the 16 lanes sharing each row set
#pragma unroll
  for (int v = 0; v < 8; ++v) {
    float m = vm[v], l = vl[v];
#pragma unroll
    for (int off = 1; off < 16; off <<= 1) {
      float om = __shfl_xor(m, off);
      float ol = __shfl_xor(l, off);
      float nm = fmaxf(m, om);
      l = l * __expf(m - nm) + ol * __expf(om - nm);
      m = nm;
    }
    vm[v] = m;
    vl[v] = 1.f / l;  // store inverse
  }

  // ---- pass 2: P = exp(S-m)/l, O += P@V, column-sum partials ----
  v8f o[4];
#pragma unroll
  for (int i = 0; i < 4; ++i) o[i] = vzero();

  for (int jc = 0; jc < 16; ++jc) {
    {
      V16 tmp;
      const unsigned short* p = qp + (size_t)(jc * 64 + qj) * 64 + qd0;
      tmp.q[0] = *(const uint4*)p;
      tmp.q[1] = *(const uint4*)(p + 8);
#pragma unroll
      for (int u = 0; u < 16; ++u) Qt[(qd0 + u) * 72 + qj] = tmp.u[u];
    }
    __syncthreads();
#pragma unroll
    for (int jp = 0; jp < 2; ++jp) {
#pragma unroll
      for (int hf = 0; hf < 2; ++hf) {
        int js = jp * 2 + hf;
        v8f s = vzero();
#pragma unroll
        for (int kc = 0; kc < 2; ++kc) {
          V16 bf;
          for (int s = 0; s < 16; ++s) bf.u[s] = Qt[(kc * 32 + (s >> 3) * 16 + hi * 8 + (s & 7)) * 72 + js * 16 + lo];
          s = wmma_bf16(af[kc], bf, s);
        }
        float csum = 0.f;
#pragma unroll
        for (int v = 0; v < 8; ++v) {
          float p = __expf(s[v] - vm[v]) * vl[v];
          csum += p;
          Ps[wave][(v + 8 * hi) * 40 + hf * 16 + lo] = f2bf(p);
        }
        csum += __shfl_xor(csum, 16);
        if (lane < 16)
          cspart[((size_t)(bh * 8 + blockIdx.x) * 8 + wave) * 1024 + jc * 64 + js * 16 + lane] = csum;
      }
      // same-wave LDS RAW: wait + compiler barrier
      asm volatile("s_wait_dscnt 0" ::: "memory");
      V16 pa;
      pa.q[0] = *(const uint4*)(&Ps[wave][lo * 40 + hi * 8]);
      pa.q[1] = *(const uint4*)(&Ps[wave][lo * 40 + 16 + hi * 8]);
#pragma unroll
      for (int dt = 0; dt < 4; ++dt) {
        V16 vf;
        for (int s = 0; s < 16; ++s) vf.u[s] = vp[(size_t)(jc * 64 + jp * 32 + (s >> 3) * 16 + hi * 8 + (s & 7)) * 64 + dt * 16 + lo];
        o[dt] = wmma_bf16(pa, vf, o[dt]);
      }
    }
    __syncthreads();
  }

  // epilogue: write O back in [b][ch][n] layout for conv_up
  const int b = bh >> 2, h = bh & 3;
#pragma unroll
  for (int dt = 0; dt < 4; ++dt) {
    int d = dt * 16 + lo;
    unsigned short* zp = z + (size_t)(b * 256 + h * 64 + d) * 1024;
#pragma unroll
    for (int v = 0; v < 8; ++v) zp[i0 + v + 8 * hi] = f2bf(o[dt][v]);
  }
}

// fixed-order reduction of column-sum partials (deterministic, no atomics)
__global__ __launch_bounds__(256) void k_colsum_reduce(const float* __restrict__ part,
                                                       float* __restrict__ colsum) {
  int idx = blockIdx.x * 256 + threadIdx.x;  // bh*1024 + j  (65536 total)
  int bh = idx >> 10, j = idx & 1023;
  const float* p = part + (size_t)bh * 64 * 1024 + j;
  float s = 0.f;
  for (int t = 0; t < 64; ++t) s += p[t * 1024];
  colsum[idx] = s;
}

// exact stable top-k by rank (matches lax.top_k ordering incl. index ties)
__global__ __launch_bounds__(256) void k_topk(const float* __restrict__ colsum,
                                              int* __restrict__ outIdx) {
  __shared__ float cs[1024];
  const int bh = blockIdx.x, tid = threadIdx.x;
  const float* p = colsum + (size_t)bh * 1024;
#pragma unroll
  for (int r = 0; r < 4; ++r) cs[r * 256 + tid] = p[r * 256 + tid];
  __syncthreads();
  for (int r = 0; r < 4; ++r) {
    int i = r * 256 + tid;
    float v = cs[i];
    int rank = 0;
    for (int j = 0; j < 1024; ++j) {
      float u = cs[j];
      rank += (u > v) || (u == v && j < i);
    }
    if (rank < KTOP) outIdx[bh * KTOP + rank] = i;
  }
}

// ---------------------------------------------------------------------------
// ConvTranspose (k=4,s=2,p=1) as 4 parity-class implicit GEMMs:
//   oh=2a+po, ow=2c+pw ; K = ic*4 + s*2 + t, ih = a+dh(po,s), iw = c+dw(pw,t)
// Same pipelined 128x128 GEMM skeleton as conv_down.
// ---------------------------------------------------------------------------

__global__ __launch_bounds__(256) void k_conv_up(
    const unsigned short* __restrict__ z,   // [16][256][1024] bf16
    const unsigned short* __restrict__ wu,  // [4][256][1024] bf16
    const float* __restrict__ bias,         // [256]
    float* __restrict__ out)                // [16][256][64][64] f32
{
  __shared__ __align__(16) unsigned short As[2][128 * 40];
  __shared__ __align__(16) unsigned short Bs[2][32 * 136];
  const int bz = blockIdx.z;
  const int b = bz >> 2, cls = bz & 3;
  const int po = cls >> 1, pw = cls & 1;
  const int nbase = blockIdx.y * 128;
  const int mbase = blockIdx.x * 128;
  const int tid = threadIdx.x;
  const int lane = tid & 31, wave = tid >> 5;
  const int hi = lane >> 4, lo = lane & 15;

  v8f acc[8];
#pragma unroll
  for (int i = 0; i < 8; ++i) acc[i] = vzero();

  const int arow = tid >> 1, acol = (tid & 1) * 16;
  const int brow = tid >> 3, bcol0 = (tid & 7) * 16;
  const unsigned short* wc = wu + (size_t)cls * 262144;

  uint4 a0, a1;
  unsigned short uv[16];
  unsigned okm;

  auto gather = [&](int kc) {
    const uint4* asrc = (const uint4*)(wc + (size_t)(mbase + arow) * 1024 + kc * 32 + acol);
    a0 = asrc[0];
    a1 = asrc[1];
    int kg = kc * 32 + brow;
    int ic = kg >> 2, s = (kg >> 1) & 1, tt = kg & 1;
    int dh = po ? (s ? 0 : 1) : (s ? -1 : 0);
    int dw = pw ? (tt ? 0 : 1) : (tt ? -1 : 0);
    const unsigned short* zp = z + (size_t)(b * 256 + ic) * 1024;
    okm = 0u;
#pragma unroll
    for (int j = 0; j < 16; ++j) {
      int n = nbase + bcol0 + j;
      int a = n >> 5, c = n & 31;
      int ih = a + dh, iw = c + dw;
      unsigned ok = ((unsigned)ih < 32u) & ((unsigned)iw < 32u);
      int hc = ok ? ih : 0, wcl = ok ? iw : 0;
      uv[j] = zp[hc * 32 + wcl];      // always-legal load, masked at stage
      okm |= ok << j;
    }
  };
  auto stage = [&](int buf) {
    uint4* ad = (uint4*)(As[buf] + arow * 40 + acol);
    ad[0] = a0;
    ad[1] = a1;
    PK8 p0, p1;
#pragma unroll
    for (int j = 0; j < 8; ++j) {
      p0.u[j] = ((okm >> j) & 1u) ? uv[j] : (unsigned short)0;
      p1.u[j] = ((okm >> (j + 8)) & 1u) ? uv[j + 8] : (unsigned short)0;
    }
    *(uint4*)(Bs[buf] + brow * 136 + bcol0) = p0.q;
    *(uint4*)(Bs[buf] + brow * 136 + bcol0 + 8) = p1.q;
  };

  gather(0);
  stage(0);
  for (int kc = 0; kc < 32; ++kc) {
    __syncthreads();
    const int cur = kc & 1;
    if (kc + 1 < 32) gather(kc + 1);
    V16 af;
    {
      int m = wave * 16 + lo;
      af.q[0] = *(const uint4*)(As[cur] + m * 40 + hi * 8);
      af.q[1] = *(const uint4*)(As[cur] + m * 40 + 16 + hi * 8);
    }
#pragma unroll
    for (int nt = 0; nt < 8; ++nt) {
      V16 bf;
      for (int s = 0; s < 16; ++s) bf.u[s] = Bs[cur][((s >> 3) * 16 + hi * 8 + (s & 7)) * 136 + nt * 16 + lo];
      acc[nt] = wmma_bf16(af, bf, acc[nt]);
    }
    if (kc + 1 < 32) stage(1 - cur);
  }
#pragma unroll
  for (int nt = 0; nt < 8; ++nt) {
    int n = nbase + nt * 16 + lo;
    int a = n >> 5, c = n & 31;
    int oh = 2 * a + po, ow = 2 * c + pw;
#pragma unroll
    for (int v = 0; v < 8; ++v) {
      int oc = mbase + wave * 16 + v + 8 * hi;
      out[((size_t)(b * 256 + oc) * 64 + oh) * 64 + ow] = acc[nt][v] + bias[oc];
    }
  }
}

// ---------------------------------------------------------------------------
// Launch
// ---------------------------------------------------------------------------

extern "C" void kernel_launch(void* const* d_in, const int* in_sizes, int n_in,
                              void* d_out, int out_size, void* d_ws, size_t ws_size,
                              hipStream_t stream) {
  (void)in_sizes; (void)n_in; (void)out_size; (void)ws_size;
  const float* x    = (const float*)d_in[0];
  const float* wdn  = (const float*)d_in[1];
  const float* bdn  = (const float*)d_in[2];
  const float* qw   = (const float*)d_in[3];
  const float* qbv  = (const float*)d_in[4];
  const float* wupw = (const float*)d_in[5];
  const float* bup  = (const float*)d_in[6];

  char* ws = (char*)d_ws;   // needs ~63.3 MB, 256B-aligned offsets
  unsigned short* wdbf   = (unsigned short*)(ws + 0);          //  2,097,152
  unsigned short* qkvwT  = (unsigned short*)(ws + 2097152);    //     24,576
  unsigned short* wubf   = (unsigned short*)(ws + 2121728);    //  2,097,152
  unsigned short* ybf    = (unsigned short*)(ws + 4218880);    //  8,388,608
  unsigned short* qbf    = (unsigned short*)(ws + 12607488);   //  8,388,608
  unsigned short* kbf    = (unsigned short*)(ws + 20996096);   //  8,388,608
  unsigned short* vbf    = (unsigned short*)(ws + 29384704);   //  8,388,608
  unsigned short* zbf    = (unsigned short*)(ws + 37773312);   //  8,388,608
  float*          cspart = (float*)(ws + 46161920);            // 16,777,216
  float*          colsum = (float*)(ws + 62939136);            //    262,144

  int*   topk = (int*)d_out;
  float* outf = (float*)d_out + BATCH * NH * KTOP;  // 16384 index slots first

  k_cvt<<<4096, 256, 0, stream>>>(wdn, wdbf, 256 * 4096);
  k_prep_qkvw<<<48, 256, 0, stream>>>(qw, qkvwT);
  k_prep_wu<<<4096, 256, 0, stream>>>(wupw, wubf);

  k_conv_down<<<dim3(2, 8, 16), 256, 0, stream>>>(x, wdbf, bdn, ybf);
  k_qkv<<<dim3(8, 64), 256, 0, stream>>>(ybf, qkvwT, qbv, qbf, kbf, vbf);
  k_attn<<<dim3(8, 64), 256, 0, stream>>>(qbf, kbf, vbf, cspart, zbf);
  k_colsum_reduce<<<256, 256, 0, stream>>>(cspart, colsum);
  k_topk<<<64, 256, 0, stream>>>(colsum, topk);
  k_conv_up<<<dim3(2, 8, 64), 256, 0, stream>>>(zbf, wubf, bup, outf);
}
